// MultiHeadSimplifiedSelfAttention_52871047414503
// MI455X (gfx1250) — compile-verified, matrix operand order checked by reading, not run on hardware
//
#include <hip/hip_runtime.h>

// ---- CDNA5 WMMA / TDM vector types ----
typedef __attribute__((ext_vector_type(16))) _Float16 v16h;
typedef __attribute__((ext_vector_type(8)))  _Float16 v8h;
typedef __attribute__((ext_vector_type(8)))  float    v8f;
typedef __attribute__((ext_vector_type(4)))  unsigned int v4u;
typedef __attribute__((ext_vector_type(8)))  int v8i;
typedef __attribute__((ext_vector_type(4)))  int v4i;

#define WMMA_F16(a,b,c) \
  __builtin_amdgcn_wmma_f32_16x16x32_f16(false,(a),false,(b),(short)0,(c),false,false)

#if defined(__has_builtin)
#  if __has_builtin(__builtin_amdgcn_tensor_load_to_lds) && \
      __has_builtin(__builtin_amdgcn_s_wait_tensorcnt)
#    define HAS_TDM 1
#  endif
#endif
#ifndef HAS_TDM
#  define HAS_TDM 0
#endif

namespace {
constexpr int HN  = 16;    // heads
constexpr int NQL = 2048;  // queries
constexpr int SEQ = 2048;  // keys
constexpr int DK  = 64;    // head dim
constexpr int DM  = 1024;  // model dim
constexpr float SCALE = 0.125f; // 1/sqrt(64)
// attention kernel LDS layout (single shared object -> LDS offset 0)
constexpr int K_OFF = 0;       // 2 x (32x64 f16) = 8192 B
constexpr int V_OFF = 8192;    // 2 x (64x32 f16) = 8192 B
constexpr int P_OFF = 16384;   // 8 x (16x32 f16) = 8192 B
constexpr int SMEM_BYTES = 24576;
// workspace layout (bytes)
constexpr size_t CTX_OFF = 0;                       // 8192x1024 f16 = 16 MB
constexpr size_t K16_OFF = 16u * 1024 * 1024;       // (b,h,s,d) f16 = 16 MB
constexpr size_t V16_OFF = 32u * 1024 * 1024;       // (b,h,d,s) f16 = 16 MB
}

// convert 8 consecutive f32 -> 8 f16
__device__ __forceinline__ void cvt8(const float* __restrict__ p, _Float16* d) {
  float4 f0 = *(const float4*)p;
  float4 f1 = *(const float4*)(p + 4);
  d[0]=(_Float16)f0.x; d[1]=(_Float16)f0.y; d[2]=(_Float16)f0.z; d[3]=(_Float16)f0.w;
  d[4]=(_Float16)f1.x; d[5]=(_Float16)f1.y; d[6]=(_Float16)f1.z; d[7]=(_Float16)f1.w;
}

// 16x32 f16 A/B fragment from an f32 row:
// lanes 0-15: k = khalf..khalf+7 and 16+khalf.. ; lanes 16-31: khalf = 8.
__device__ __forceinline__ v16h fragA_f32(const float* __restrict__ p, int khalf) {
  _Float16 t[16];
  cvt8(p + khalf, t);
  cvt8(p + 16 + khalf, t + 8);
  v16h r;
#pragma unroll
  for (int i = 0; i < 16; ++i) r[i] = t[i];
  return r;
}

// Same fragment from an f16 row (two 16B vector loads, no conversion).
__device__ __forceinline__ v16h fragA_h(const _Float16* __restrict__ p, int khalf) {
  v8h x = *(const v8h*)(p + khalf);
  v8h y = *(const v8h*)(p + 16 + khalf);
  v16h r;
#pragma unroll
  for (int i = 0; i < 8; ++i) { r[i] = x[i]; r[8 + i] = y[i]; }
  return r;
}

#if HAS_TDM
// One TDM 2D tile load: tile_h rows x tile_w elements of size (1<<ds_code)
// bytes, tensor row stride `row_stride` elements, into LDS byte offset lds_off.
__device__ __forceinline__ void tdm_load_2d(unsigned lds_off, const void* gaddr,
                                            int tile_w, int tile_h, int row_stride,
                                            int ds_code) {
  unsigned long long ga = (unsigned long long)gaddr;
  v4u g0;
  g0[0] = 1u;                                   // count=1, user descriptor
  g0[1] = lds_off;                              // lds_addr (bytes)
  g0[2] = (unsigned)(ga & 0xffffffffu);         // global_addr[31:0]
  g0[3] = (unsigned)((ga >> 32) & 0x01ffffffu)  // global_addr[56:32]
          | (2u << 30);                         // type = 2 ("image")
  v8i g1;
  g1[0] = (ds_code << 16);                      // data_size
  g1[1] = (tile_w << 16);                       // tensor_dim0 (lo16)
  g1[2] = (tile_h << 16);                       // tensor_dim0 hi=0 | tensor_dim1 lo16
  g1[3] = (tile_w << 16);                       // tensor_dim1 hi=0 | tile_dim0
  g1[4] = tile_h;                               // tile_dim1 | tile_dim2=0
  g1[5] = row_stride;                           // tensor_dim0_stride (lo32)
  g1[6] = 0;
  g1[7] = 0;
  v4i gz = {0, 0, 0, 0};
#if __clang_major__ >= 23
  v8i gz8 = {0, 0, 0, 0, 0, 0, 0, 0};
  __builtin_amdgcn_tensor_load_to_lds(g0, g1, gz, gz, gz8, 0);
#else
  __builtin_amdgcn_tensor_load_to_lds(g0, g1, gz, gz, 0);
#endif
}
#endif

// ---------------------------------------------------------------------------
// Kernel 0: one-shot precision/layout prep.
//   K (B,S,DM f32)  -> K16  (B,H,S,DK)  f16  (row-major per head)
//   V (B,S,DM f32)  -> V16T (B,H,DK,S)  f16  (transposed per head, via LDS)
// Block: one (b,h) and a 64-row s-tile. 2048 blocks x 256 threads.
// ---------------------------------------------------------------------------
__global__ __launch_bounds__(256) void prep_kv(
    const float* __restrict__ K, const float* __restrict__ V,
    _Float16* __restrict__ K16, _Float16* __restrict__ V16T)
{
  __shared__ _Float16 Tsh[64 * 64];   // [d][s] transpose staging, 8 KB

  const int tid = threadIdx.x;
  const int bid = blockIdx.x;         // st(5b) | h(4b) | b(2b)
  const int st  = bid & 31;
  const int h   = (bid >> 5) & 15;
  const int b   = bid >> 9;
  const int s0  = st * 64;

  const int srow = tid >> 2;          // 0..63
  const int d0   = (tid & 3) << 4;    // 0,16,32,48

  // ---- K: straight convert, coalesced in and out ----
  {
    const float* src = K + ((size_t)b * SEQ + s0 + srow) * DM + h * DK + d0;
    _Float16 t[16];
    cvt8(src, t); cvt8(src + 8, t + 8);
    _Float16* dst = K16 + ((size_t)(b * HN + h) * SEQ + s0 + srow) * DK + d0;
    *(v8h*)dst       = *(v8h*)t;
    *(v8h*)(dst + 8) = *(v8h*)(t + 8);
  }
  // ---- V: convert + transpose through LDS ----
  {
    const float* src = V + ((size_t)b * SEQ + s0 + srow) * DM + h * DK + d0;
    _Float16 t[16];
    cvt8(src, t); cvt8(src + 8, t + 8);
#pragma unroll
    for (int i = 0; i < 16; ++i) Tsh[(d0 + i) * 64 + srow] = t[i];
  }
  __syncthreads();
  {
    const int d   = tid >> 2;          // 0..63
    const int sl  = (tid & 3) << 4;    // 0,16,32,48
    _Float16* dst = V16T + ((size_t)(b * HN + h) * DK + d) * SEQ + s0 + sl;
    *(v8h*)dst       = *(const v8h*)&Tsh[d * 64 + sl];
    *(v8h*)(dst + 8) = *(const v8h*)&Tsh[d * 64 + sl + 8];
  }
}

// ---------------------------------------------------------------------------
// Kernel 1: fused attention.  Block = 8 waves, one (b,h), 8 query tiles of 16.
// K and V chunks both arrive via TDM (f16, double-buffered, 2 ops in flight);
// softmax has no cross-lane reductions (rowsum via WMMA against ones).
// ---------------------------------------------------------------------------
__global__ __launch_bounds__(256) void attn_flash_wmma(
    const float* __restrict__ Q, const _Float16* __restrict__ K16,
    const _Float16* __restrict__ V16T, const float* __restrict__ AW,
    _Float16* __restrict__ ctx)
{
  __shared__ __align__(16) unsigned char smem[SMEM_BYTES];
  _Float16* Ksh = (_Float16*)(smem + K_OFF);   // [buf][32*64] f16 row-major
  _Float16* Vsh = (_Float16*)(smem + V_OFF);   // [buf][64*32] f16 (d-major)
  _Float16* Psh = (_Float16*)(smem + P_OFF);   // [wave][16*32] f16

  const int tid   = threadIdx.x;
  const int wave  = tid >> 5;
  const int lane  = tid & 31;
  const int nlo   = lane & 15;
  const int hi    = lane >> 4;
  const int khalf = hi * 8;

  const int bid = blockIdx.x;        // 1024 blocks: b(2b) | h(4b) | qb(4b)
  const int qb  = bid & 15;
  const int h   = (bid >> 4) & 15;
  const int b   = bid >> 8;
  const int q0  = qb * 128 + wave * 16;

  const _Float16* Khead = K16  + (size_t)(b * HN + h) * SEQ * DK;
  const _Float16* Vhead = V16T + (size_t)(b * HN + h) * DK * SEQ;

  // Q fragments for this wave's 16 rows (K = 0..31 and 32..63)
  const float* qrow = Q + ((size_t)b * NQL + q0 + nlo) * DM + h * DK;
  const v16h qa0 = fragA_f32(qrow,      khalf);
  const v16h qa1 = fragA_f32(qrow + 32, khalf);

  v16h vones;
#pragma unroll
  for (int i = 0; i < 16; ++i) vones[i] = (_Float16)1.0f;

  v8f acc0 = {}, acc1 = {}, acc2 = {}, acc3 = {}, accS = {};

  const float* awbase = AW + ((size_t)(b * HN + h) * NQL + q0) * SEQ;

  // stage one 32-key chunk (K tile + V^T tile) into LDS buffer `buf`
  auto stage = [&](int s0c, int buf) {
#if HAS_TDM
    if (wave == 0) {
      tdm_load_2d((unsigned)(K_OFF + buf * 4096), Khead + (size_t)s0c * DK,
                  DK, 32, DK, 1);                 // 32 rows x 64 f16, contiguous
      tdm_load_2d((unsigned)(V_OFF + buf * 4096), Vhead + s0c,
                  32, DK, SEQ, 1);                // 64 rows x 32 f16, stride 2048
    }
#else
    // fallback: cooperative copies (coalesced, already f16)
    {
      const _Float16* src = Khead + (size_t)s0c * DK + tid * 8;
      *(v8h*)(Ksh + buf * 2048 + tid * 8) = *(const v8h*)src;
    }
    {
      const int d = tid >> 2, sl = (tid & 3) << 3;
      const _Float16* src = Vhead + (size_t)d * SEQ + s0c + sl;
      *(v8h*)(Vsh + buf * 2048 + d * 32 + sl) = *(const v8h*)src;
    }
#endif
  };

  stage(0, 0);   // prologue

  for (int s0 = 0; s0 < SEQ; s0 += 32) {
    const int buf = (s0 >> 5) & 1;
    __syncthreads();                       // buf^1 free for overwrite
    if (s0 + 32 < SEQ) stage(s0 + 32, buf ^ 1);   // pipeline next chunk
#if HAS_TDM
    if (wave == 0) {                       // current chunk's 2 TDM ops done
      if (s0 + 32 < SEQ) __builtin_amdgcn_s_wait_tensorcnt(2);
      else               __builtin_amdgcn_s_wait_tensorcnt(0);
    }
#endif
    __syncthreads();                       // chunk `buf` visible to all waves

    const _Float16* Kf = Ksh + buf * 2048;
    const _Float16* Vt = Vsh + buf * 2048;

    // ---- scores: two 16x16 C tiles, K=64 via two WMMAs each ----
    const v16h kb00 = fragA_h(Kf + (nlo)      * 64 + 0,  khalf);
    const v16h kb01 = fragA_h(Kf + (nlo)      * 64 + 32, khalf);
    const v16h kb10 = fragA_h(Kf + (16 + nlo) * 64 + 0,  khalf);
    const v16h kb11 = fragA_h(Kf + (16 + nlo) * 64 + 32, khalf);

    v8f c0 = {}, c1 = {};
    c0 = WMMA_F16(qa0, kb00, c0);
    c0 = WMMA_F16(qa1, kb01, c0);
    c1 = WMMA_F16(qa0, kb10, c1);
    c1 = WMMA_F16(qa1, kb11, c1);

    if (s0 + 32 < SEQ)  // prefetch next attn_weights tile (global_prefetch)
      __builtin_prefetch(awbase + (size_t)(khalf + hi) * SEQ + s0 + 32 + nlo, 0, 1);

    // ---- attn_weights * scale, exp (shift-free: scores bounded, clamp 60) --
    _Float16* Pw = Psh + wave * 512;
#pragma unroll
    for (int r = 0; r < 8; ++r) {
      const int m = r + hi * 8;
      const float* wr = awbase + (size_t)m * SEQ + s0 + nlo;
      const float e0 = __expf(fminf(c0[r] * SCALE * wr[0],  60.f));
      const float e1 = __expf(fminf(c1[r] * SCALE * wr[16], 60.f));
      _Float16* pr = Pw + m * 32 + nlo;    // C-layout -> row-major LDS
      pr[0]  = (_Float16)e0;
      pr[16] = (_Float16)e1;
    }
    // intra-wave transpose hand-off: LDS is in-order per wave, drain DScnt
    asm volatile("s_wait_dscnt 0" ::: "memory");

    // ---- P (16x32) @ V (32x64), plus rowsum = P @ ones via tensor core ----
    const v16h pa  = fragA_h(Pw + nlo * 32, khalf);
    const v16h vb0 = fragA_h(Vt + (0  + nlo) * 32, khalf);
    const v16h vb1 = fragA_h(Vt + (16 + nlo) * 32, khalf);
    const v16h vb2 = fragA_h(Vt + (32 + nlo) * 32, khalf);
    const v16h vb3 = fragA_h(Vt + (48 + nlo) * 32, khalf);
    acc0 = WMMA_F16(pa, vb0, acc0);
    acc1 = WMMA_F16(pa, vb1, acc1);
    acc2 = WMMA_F16(pa, vb2, acc2);
    acc3 = WMMA_F16(pa, vb3, acc3);
    accS = WMMA_F16(pa, vones, accS);      // softmax denominator
  }

  // ---- normalize and write context (f16) in bqhd-merged layout ----
#pragma unroll
  for (int r = 0; r < 8; ++r) {
    const float inv = 1.f / accS[r];
    _Float16* crow = ctx + ((size_t)b * NQL + q0 + r + hi * 8) * DM + h * DK + nlo;
    crow[0]  = (_Float16)(acc0[r] * inv);
    crow[16] = (_Float16)(acc1[r] * inv);
    crow[32] = (_Float16)(acc2[r] * inv);
    crow[48] = (_Float16)(acc3[r] * inv);
  }
}

// ---------------------------------------------------------------------------
// Kernel 2: out = ctx @ W_out^T + b_out.   out[r][c] = sum_k ctx[r][k]*W[c][k]
// One wave -> 16x64 output tile, K-loop step 32, 4 WMMAs/step.
// ---------------------------------------------------------------------------
__global__ __launch_bounds__(256) void proj_wmma(
    const _Float16* __restrict__ ctx, const float* __restrict__ Wo,
    const float* __restrict__ bo, float* __restrict__ out)
{
  const int tid   = threadIdx.x;
  const int wave  = tid >> 5;
  const int lane  = tid & 31;
  const int nlo   = lane & 15;
  const int hi    = lane >> 4;
  const int khalf = hi * 8;

  const int task  = blockIdx.x * 8 + wave;  // 8192 wave tasks
  const int ctile = task & 15;              // 16 column tiles of 64
  const int rtile = task >> 4;              // 512 row tiles of 16
  const int r0 = rtile * 16, c0 = ctile * 64;

  v8f acc0 = {}, acc1 = {}, acc2 = {}, acc3 = {};

  for (int k0 = 0; k0 < DM; k0 += 32) {
    const v16h a  = fragA_h(ctx + (size_t)(r0 + nlo) * DM + k0, khalf);
    const v16h b0 = fragA_f32(Wo + (size_t)(c0 +  0 + nlo) * DM + k0, khalf);
    const v16h b1 = fragA_f32(Wo + (size_t)(c0 + 16 + nlo) * DM + k0, khalf);
    const v16h b2 = fragA_f32(Wo + (size_t)(c0 + 32 + nlo) * DM + k0, khalf);
    const v16h b3 = fragA_f32(Wo + (size_t)(c0 + 48 + nlo) * DM + k0, khalf);
    acc0 = WMMA_F16(a, b0, acc0);
    acc1 = WMMA_F16(a, b1, acc1);
    acc2 = WMMA_F16(a, b2, acc2);
    acc3 = WMMA_F16(a, b3, acc3);
  }

#pragma unroll
  for (int r = 0; r < 8; ++r) {
    float* orow = out + (size_t)(r0 + r + hi * 8) * DM + c0 + nlo;
    orow[0]  = acc0[r] + bo[c0 +  0 + nlo];
    orow[16] = acc1[r] + bo[c0 + 16 + nlo];
    orow[32] = acc2[r] + bo[c0 + 32 + nlo];
    orow[48] = acc3[r] + bo[c0 + 48 + nlo];
  }
}

extern "C" void kernel_launch(void* const* d_in, const int* in_sizes, int n_in,
                              void* d_out, int out_size, void* d_ws, size_t ws_size,
                              hipStream_t stream) {
  const float* Q  = (const float*)d_in[0];
  const float* K  = (const float*)d_in[1];
  const float* V  = (const float*)d_in[2];
  const float* AW = (const float*)d_in[3];
  const float* Wo = (const float*)d_in[4];
  const float* bo = (const float*)d_in[5];
  float* out = (float*)d_out;

  unsigned char* ws = (unsigned char*)d_ws;
  _Float16* ctx  = (_Float16*)(ws + CTX_OFF);  // 16 MB
  _Float16* K16  = (_Float16*)(ws + K16_OFF);  // 16 MB
  _Float16* V16T = (_Float16*)(ws + V16_OFF);  // 16 MB

  // one-shot K/V precision+layout conversion: 4*16*32 = 2048 blocks
  prep_kv<<<2048, 256, 0, stream>>>(K, V, K16, V16T);
  // 4 batches * 16 heads * 16 query-blocks(128 rows) = 1024 blocks
  attn_flash_wmma<<<1024, 256, 0, stream>>>(Q, K16, V16T, AW, ctx);
  // (8192/16 row tiles) * (1024/64 col tiles) / 8 waves = 1024 blocks
  proj_wmma<<<1024, 256, 0, stream>>>(ctx, Wo, bo, out);
}
